// MLPAttention_51642686767293
// MI455X (gfx1250) — compile-verified, matrix operand order checked by reading, not run on hardware
//
#include <hip/hip_runtime.h>
#include <hip/hip_bf16.h>
#include <math.h>

#define BB   16
#define LL   2048
#define DD   1024
#define QQ   1024
#define HH   256
#define NEG_NUM -10000.0f

typedef __attribute__((ext_vector_type(16))) __bf16 v16bf;
typedef __attribute__((ext_vector_type(8)))  float  v8f;

// ---------------------------------------------------------------------------
// Hardware tanh: CDNA5 has V_TANH_F32 (TRANS pipe, co-executes with WMMA).
// Prefer the clang builtin; fall back to inline asm. For raw asm we add
// v_nops per the ISA TRANS-op RAW hazard rule (1 indep op before use).
// ---------------------------------------------------------------------------
__device__ __forceinline__ float fast_tanh(float x) {
#if __has_builtin(__builtin_amdgcn_tanhf)
    return __builtin_amdgcn_tanhf(x);
#else
    float y;
    asm volatile("v_tanh_f32 %0, %1\n\tv_nop\n\tv_nop" : "=v"(y) : "v"(x));
    return y;
#endif
}

// ---------------------------------------------------------------------------
// Kernel 1: w2 = g * v / ||v||   (single block, 256 threads)
// ---------------------------------------------------------------------------
__global__ __launch_bounds__(256)
void w2_kernel(const float* __restrict__ v, const float* __restrict__ g,
               float* __restrict__ w2)
{
    __shared__ float red[256];
    const int h = threadIdx.x;
    const float vv = v[h];
    red[h] = vv * vv;
    __syncthreads();
    for (int off = 128; off > 0; off >>= 1) {
        if (h < off) red[h] += red[h + off];
        __syncthreads();
    }
    const float norm = sqrtf(red[0]);
    w2[h] = g[0] * vv / norm;
}

// ---------------------------------------------------------------------------
// Kernel 2: qproj[b,h] = b1[h] + query[b,:] . W1[h, D:D+Q]
// grid = (B), block = 256 (= H)
// ---------------------------------------------------------------------------
__global__ __launch_bounds__(256)
void qproj_kernel(const float* __restrict__ query, const float* __restrict__ W1,
                  const float* __restrict__ b1, float* __restrict__ qproj)
{
    const int b = blockIdx.x;
    const int h = threadIdx.x;
    const float* __restrict__ wq = W1 + (size_t)h * (DD + QQ) + DD;
    const float* __restrict__ q  = query + (size_t)b * QQ;
    float acc = b1[h];
    for (int k = 0; k < QQ; k += 4) {
        const float4 qa = *(const float4*)(q + k);
        const float4 wa = *(const float4*)(wq + k);
        acc += qa.x * wa.x + qa.y * wa.y + qa.z * wa.z + qa.w * wa.w;
    }
    qproj[b * HH + h] = acc;
}

// ---------------------------------------------------------------------------
// Kernel 3: fused score GEMM via v_wmma_f32_16x16x32_bf16
//   score[b,l] = sum_h tanh( ctx[b,l,:].Wc[h,:] + qproj[b,h] ) * w2[h] + b2
// grid = (L/64, B), block = 128 (4 waves, 16 rows of L each)
// ---------------------------------------------------------------------------
__global__ __launch_bounds__(128)
void score_wmma_kernel(const float* __restrict__ ctx, const float* __restrict__ W1,
                       const float* __restrict__ qproj, const float* __restrict__ w2,
                       const float* __restrict__ b2, float* __restrict__ score)
{
    const int b    = blockIdx.y;
    const int wave = threadIdx.x >> 5;
    const int lane = threadIdx.x & 31;
    const int half = lane >> 4;          // K-half selector (ISA A/B bf16 layout)
    const int l16  = lane & 15;          // A: row M;  B: column N (= Wc row h)
    const int m0   = blockIdx.x * 64 + wave * 16;

    // A operand: ctx row (m0 + l16), K chunks at half*8 and half*8+16
    const float* __restrict__ arow =
        ctx + ((size_t)b * LL + (m0 + l16)) * DD + half * 8;

    float partial[8];
#pragma unroll
    for (int r = 0; r < 8; ++r) partial[r] = 0.0f;

    for (int ht = 0; ht < HH / 16; ++ht) {
        // B operand: Wc row (ht*16 + l16) — W1 row stride is D+Q
        const float* __restrict__ brow =
            W1 + (size_t)(ht * 16 + l16) * (DD + QQ) + half * 8;

        // prefetch next ht's Wc row into the cache hierarchy (global_prefetch_b8)
        if (ht + 1 < HH / 16)
            __builtin_prefetch(brow + (size_t)16 * (DD + QQ), 0, 1);

        v8f acc = {};
        for (int k0 = 0; k0 < DD; k0 += 32) {
            const v8f alo = *(const v8f*)(arow + k0);
            const v8f ahi = *(const v8f*)(arow + k0 + 16);
            const v8f blo = *(const v8f*)(brow + k0);
            const v8f bhi = *(const v8f*)(brow + k0 + 16);
            v16bf A, Bm;
#pragma unroll
            for (int i = 0; i < 8; ++i) {
                A[i]      = (__bf16)alo[i];
                A[i + 8]  = (__bf16)ahi[i];
                Bm[i]     = (__bf16)blo[i];
                Bm[i + 8] = (__bf16)bhi[i];
            }
            // D = A(16x32) * B(32x16) + C, f32 accumulate
            acc = __builtin_amdgcn_wmma_f32_16x16x32_bf16(
                false, A, false, Bm, (short)0, acc, false, false);
        }

        // C/D layout: VGPR r, lanes 0-15 -> M=r, lanes 16-31 -> M=r+8; N = lane&15
        const int h    = ht * 16 + l16;
        const float qv = qproj[b * HH + h];
        const float wv = w2[h];
#pragma unroll
        for (int r = 0; r < 8; ++r)
            partial[r] += fast_tanh(acc[r] + qv) * wv;
    }

    // Reduce over N (16 lanes per half; xor masks < 16 stay inside each half)
#pragma unroll
    for (int r = 0; r < 8; ++r) {
        float p = partial[r];
        p += __shfl_xor(p, 1, 32);
        p += __shfl_xor(p, 2, 32);
        p += __shfl_xor(p, 4, 32);
        p += __shfl_xor(p, 8, 32);
        partial[r] = p;
    }

    if (l16 == 0) {
        const float bb = b2[0];
#pragma unroll
        for (int r = 0; r < 8; ++r)
            score[(size_t)b * LL + m0 + r + 8 * half] = partial[r] + bb;
    }
}

// ---------------------------------------------------------------------------
// Kernel 4: masked softmax over L per batch. grid = (B), block = 256
// ---------------------------------------------------------------------------
__global__ __launch_bounds__(256)
void softmax_kernel(const float* __restrict__ score, const int* __restrict__ mask,
                    float* __restrict__ p_ctx)
{
    __shared__ float red[256];
    const int b   = blockIdx.x;
    const int tid = threadIdx.x;

    float s[LL / 256];
    float mx = -INFINITY;
#pragma unroll
    for (int i = 0; i < LL / 256; ++i) {
        const int l = tid + i * 256;
        float v = score[(size_t)b * LL + l];
        if (mask[b * LL + l] == 1) v = NEG_NUM;
        s[i] = v;
        mx = fmaxf(mx, v);
    }
    red[tid] = mx;
    __syncthreads();
    for (int off = 128; off > 0; off >>= 1) {
        if (tid < off) red[tid] = fmaxf(red[tid], red[tid + off]);
        __syncthreads();
    }
    mx = red[0];
    __syncthreads();

    float sum = 0.0f;
#pragma unroll
    for (int i = 0; i < LL / 256; ++i) {
        s[i] = __expf(s[i] - mx);
        sum += s[i];
    }
    red[tid] = sum;
    __syncthreads();
    for (int off = 128; off > 0; off >>= 1) {
        if (tid < off) red[tid] += red[tid + off];
        __syncthreads();
    }
    const float inv = 1.0f / red[0];
#pragma unroll
    for (int i = 0; i < LL / 256; ++i)
        p_ctx[(size_t)b * LL + tid + i * 256] = s[i] * inv;
}

// ---------------------------------------------------------------------------
// Kernel 5: expected_ctx[b,d] = sum_l p[b,l] * ctx[b,l,d]
// grid = (D/256, B), block = 256 — coalesced over d, ctx re-read hits L2
// ---------------------------------------------------------------------------
__global__ __launch_bounds__(256)
void ectx_kernel(const float* __restrict__ ctx, const float* __restrict__ p,
                 float* __restrict__ out)
{
    const int b = blockIdx.y;
    const int d = blockIdx.x * 256 + threadIdx.x;
    const float* __restrict__ base = ctx + (size_t)b * LL * DD + d;
    const float* __restrict__ pb   = p + (size_t)b * LL;
    float acc0 = 0.0f, acc1 = 0.0f, acc2 = 0.0f, acc3 = 0.0f;
    for (int l = 0; l < LL; l += 4) {
        const float4 pv = *(const float4*)(pb + l);
        acc0 += pv.x * base[(size_t)(l + 0) * DD];
        acc1 += pv.y * base[(size_t)(l + 1) * DD];
        acc2 += pv.z * base[(size_t)(l + 2) * DD];
        acc3 += pv.w * base[(size_t)(l + 3) * DD];
    }
    out[(size_t)b * DD + d] = (acc0 + acc1) + (acc2 + acc3);
}

// ---------------------------------------------------------------------------
extern "C" void kernel_launch(void* const* d_in, const int* in_sizes, int n_in,
                              void* d_out, int out_size, void* d_ws, size_t ws_size,
                              hipStream_t stream)
{
    const float* ctx   = (const float*)d_in[0];   // (B, L, D)
    const float* query = (const float*)d_in[1];   // (B, Q)
    const int*   mask  = (const int*)  d_in[2];   // (B, L)
    const float* W1    = (const float*)d_in[3];   // (H, D+Q)
    const float* b1    = (const float*)d_in[4];   // (H,)
    const float* v     = (const float*)d_in[5];   // (1, H)
    const float* g     = (const float*)d_in[6];   // (1, 1)
    const float* b2    = (const float*)d_in[7];   // (1,)

    float* out_ectx = (float*)d_out;              // (B, D)
    float* out_p    = (float*)d_out + BB * DD;    // (B, L)

    float* ws    = (float*)d_ws;
    float* w2    = ws;                            // H
    float* qproj = ws + HH;                       // B*H
    float* score = ws + HH + BB * HH;             // B*L

    w2_kernel<<<1, 256, 0, stream>>>(v, g, w2);
    qproj_kernel<<<BB, 256, 0, stream>>>(query, W1, b1, qproj);

    dim3 sgrid(LL / 64, BB);
    score_wmma_kernel<<<sgrid, 128, 0, stream>>>(ctx, W1, qproj, w2, b2, score);

    softmax_kernel<<<BB, 256, 0, stream>>>(score, mask, out_p);

    dim3 egrid(DD / 256, BB);
    ectx_kernel<<<egrid, 256, 0, stream>>>(ctx, out_p, out_ectx);
}
